// GraphMDN_43121471652445
// MI455X (gfx1250) — matrix operand order repaired and checked by reading, not compiled, not run on hardware
//
#include <hip/hip_runtime.h>
#include <hip/hip_bf16.h>

// ---------------------------------------------------------------------------
// GraphMDN on MI455X (gfx1250): 2x GCNConv (WMMA bf16 GEMM + f32 atomic
// scatter over edges) + MDN heads (WMMA GEMM + softmax).
// Weight tiles staged to LDS with gfx1250 async global->LDS copies.
// ---------------------------------------------------------------------------

typedef __attribute__((ext_vector_type(16))) __bf16 v16bf;
typedef __attribute__((ext_vector_type(8)))  float  v8f;

union FragAB { v16bf v; unsigned u[8]; };
union FragC  { v8f   v; float    f[8]; };

static __device__ __forceinline__ unsigned short f32_to_bf16(float f) {
  union { float f; unsigned u; } x; x.f = f;
  unsigned r = x.u + 0x7FFFu + ((x.u >> 16) & 1u);   // round-to-nearest-even
  return (unsigned short)(r >> 16);
}

// ---------------------------------------------------------------------------
// Weight prep: W1,W2 (f32 [k][n]) -> bf16 [n][k]; head weights packed into
// one [64][128] bf16 matrix (cols 0-19 pi, 20-39 mu, 40-59 ls, 60-63 zero).
// ---------------------------------------------------------------------------
__global__ void prep_weights(const float* __restrict__ W1,
                             const float* __restrict__ W2,
                             const float* __restrict__ piW,
                             const float* __restrict__ muW,
                             const float* __restrict__ lsW,
                             const float* __restrict__ pib,
                             const float* __restrict__ mub,
                             const float* __restrict__ lsb,
                             unsigned short* __restrict__ Wt1,
                             unsigned short* __restrict__ Wt2,
                             unsigned short* __restrict__ Wht,
                             float* __restrict__ bh) {
  const int idx = blockIdx.x * blockDim.x + threadIdx.x;
  if (idx < 16384) {
    const int n = idx >> 7, k = idx & 127;
    Wt1[n * 128 + k] = f32_to_bf16(W1[k * 128 + n]);
  } else if (idx < 32768) {
    const int j = idx - 16384;
    const int n = j >> 7, k = j & 127;
    Wt2[n * 128 + k] = f32_to_bf16(W2[k * 128 + n]);
  } else if (idx < 32768 + 8192) {
    const int j = idx - 32768;
    const int n = j >> 7, k = j & 127;
    float v = 0.f;
    if (n < 20)      v = piW[k * 20 + n];
    else if (n < 40) v = muW[k * 20 + (n - 20)];
    else if (n < 60) v = lsW[k * 20 + (n - 40)];
    Wht[n * 128 + k] = f32_to_bf16(v);
  } else if (idx < 32768 + 8192 + 64) {
    const int n = idx - 40960;
    float v = 0.f;
    if (n < 20)      v = pib[n];
    else if (n < 40) v = mub[n - 20];
    else if (n < 60) v = lsb[n - 40];
    bh[n] = v;
  }
}

__global__ void zero_f32(float* __restrict__ p, int n) {
  const int i = blockIdx.x * blockDim.x + threadIdx.x;
  if (i < n) p[i] = 0.f;
}

__global__ void edge_deg(const int* __restrict__ dst, float* __restrict__ deg,
                         int E) {
  const int e = blockIdx.x * blockDim.x + threadIdx.x;
  if (e < E) unsafeAtomicAdd(&deg[dst[e]], 1.0f);
}

__global__ void inv_sqrt_deg(float* __restrict__ d, int n) {
  const int i = blockIdx.x * blockDim.x + threadIdx.x;
  if (i < n) d[i] = rsqrtf(d[i] + 1.0f);       // self-loop adds 1 to degree
}

// ---------------------------------------------------------------------------
// GEMM: Out[M x NOUT] = X[M x 128] (f32 -> bf16) @ Wt^T, Wt bf16 stored
// [n][k].  4 waves per block; each wave owns a 16-row slice of a 64-row M
// tile.  Weights staged to LDS via async global->LDS b128 copies (ASYNCcnt);
// X tile converted f32->bf16 by VALU in parallel.  A/B fragments use contiguous
// k = 16*half + s in both A and B.
// ---------------------------------------------------------------------------
template <int NOUT>
__global__ __launch_bounds__(128) void gemm_bf16_wmma(
    const float* __restrict__ X, const unsigned short* __restrict__ Wt,
    const float* __restrict__ bias, float* __restrict__ Out, int M) {
  __shared__ unsigned short Al[64 * 128];      // 16 KB  X tile (bf16)
  __shared__ unsigned short Bl[NOUT * 128];    // 32/16 KB weights (bf16 [n][k])

  const int tid  = threadIdx.x;
  const int lane = tid & 31;
  const int wave = tid >> 5;        // 0..3
  const int hi   = lane >> 4;       // half-wave select
  const int l15  = lane & 15;
  const int rowBase = blockIdx.x * 64;

  // ---- Stage weights: async global->LDS (tracked by ASYNCcnt). Each lane
  // copies 16B per issue; 128 threads move 2 KB per iteration.
  {
    constexpr int WBYTES = NOUT * 256;          // NOUT*128 bf16
    unsigned lds = (unsigned)(uintptr_t)(&Bl[0]) + tid * 16;  // LDS = flat[31:0]
    unsigned long long g = (unsigned long long)(uintptr_t)Wt + tid * 16;
#pragma unroll
    for (int off = 0; off < WBYTES; off += 2048) {
      asm volatile("global_load_async_to_lds_b128 %0, %1, off"
                   :: "v"(lds), "v"(g) : "memory");
      lds += 2048;
      g   += 2048;
    }
  }

  // ---- Stage X tile with f32 -> bf16 conversion (64 rows x 128 cols)
  for (int i = tid; i < 2048; i += 128) {
    const int flat = i << 2;
    const int r = flat >> 7, c = flat & 127;
    const int row = rowBase + r;
    float4 v = make_float4(0.f, 0.f, 0.f, 0.f);
    if (row < M) v = *(const float4*)(X + (size_t)row * 128 + c);
    unsigned short* p = &Al[r * 128 + c];
    p[0] = f32_to_bf16(v.x); p[1] = f32_to_bf16(v.y);
    p[2] = f32_to_bf16(v.z); p[3] = f32_to_bf16(v.w);
  }
  asm volatile("s_wait_asynccnt 0x0" ::: "memory");
  __syncthreads();

  constexpr int NT = NOUT >> 4;
  const int arow = (wave << 4) + l15;           // A row for this lane
  const bool full = (rowBase + 64) <= M;        // wave-uniform tail check
  const int rbase = rowBase + (wave << 4) + (hi << 3);

#pragma unroll
  for (int t = 0; t < NT; ++t) {
    FragC acc;
#pragma unroll
    for (int i = 0; i < 8; ++i) acc.f[i] = 0.f;
    const int col = (t << 4) + l15;
#pragma unroll
    for (int kc = 0; kc < 4; ++kc) {            // K = 4 x 32
      FragAB a, b;
      // A 16x32 bf16: lanes 0-15 -> K {0..15}; lanes 16-31 -> {16..31}
      const unsigned* Ap = (const unsigned*)&Al[arow * 128 + kc * 32];
#pragma unroll
      for (int j = 0; j < 4; ++j) {
        a.u[j]     = Ap[hi * 8 + j];
        a.u[4 + j] = Ap[hi * 8 + 4 + j];
      }
      // B 32x16 bf16: lane = column, K contiguous per half-wave
      const unsigned* Bp = (const unsigned*)&Bl[col * 128 + kc * 32 + hi * 16];
#pragma unroll
      for (int j = 0; j < 8; ++j) b.u[j] = Bp[j];

      acc.v = __builtin_amdgcn_wmma_f32_16x16x32_bf16(
          false, a.v, false, b.v, (short)0, acc.v, false, false);
    }
    // C/D layout: VGPR r -> row (r + 8*hi), lane&15 -> column
    const float badd = bias ? bias[col] : 0.f;
    if (full) {
#pragma unroll
      for (int r = 0; r < 8; ++r)
        Out[(size_t)(rbase + r) * NOUT + col] = acc.f[r] + badd;
    } else {
#pragma unroll
      for (int r = 0; r < 8; ++r) {
        const int row = rbase + r;
        if (row < M) Out[(size_t)row * NOUT + col] = acc.f[r] + badd;
      }
    }
  }
}

// ---------------------------------------------------------------------------
// Edge scatter: one wave32 per edge; each lane handles 4 channels (float4
// gather, 4x global f32 atomic adds into destination row).
// ---------------------------------------------------------------------------
__global__ __launch_bounds__(256) void scatter_edges(
    const int* __restrict__ src, const int* __restrict__ dst,
    const float* __restrict__ H, const float* __restrict__ dis,
    float* __restrict__ Agg, int E) {
  const int g = blockIdx.x * 256 + threadIdx.x;
  const int e = g >> 5;
  if (e >= E) return;
  const int lane = threadIdx.x & 31;
  const int s = src[e], d = dst[e];
  const float norm = dis[s] * dis[d];
  const float4 hv = *(const float4*)(H + (size_t)s * 128 + lane * 4);
  float* out = Agg + (size_t)d * 128 + lane * 4;
  unsafeAtomicAdd(out + 0, hv.x * norm);
  unsafeAtomicAdd(out + 1, hv.y * norm);
  unsafeAtomicAdd(out + 2, hv.z * norm);
  unsafeAtomicAdd(out + 3, hv.w * norm);
}

// agg <- relu(agg + h * dis^2 + b), in place
__global__ void finalize_relu(float* __restrict__ Agg,
                              const float* __restrict__ H,
                              const float* __restrict__ dis,
                              const float* __restrict__ b, int N) {
  const int idx = blockIdx.x * blockDim.x + threadIdx.x;
  if (idx >= N * 128) return;
  const int i = idx >> 7;
  const int c = idx & 127;
  const float di = dis[i];
  const float v = Agg[idx] + H[idx] * di * di + b[c];
  Agg[idx] = v > 0.f ? v : 0.f;
}

// softmax(pi) + split heads into concatenated output
__global__ void heads_out(const float* __restrict__ Hd,
                          float* __restrict__ out, int N) {
  const int i = blockIdx.x * blockDim.x + threadIdx.x;
  if (i >= N) return;
  const float* h = Hd + (size_t)i * 64;
  float m = h[0];
#pragma unroll
  for (int k = 1; k < 20; ++k) m = fmaxf(m, h[k]);
  float ex[20]; float s = 0.f;
#pragma unroll
  for (int k = 0; k < 20; ++k) { ex[k] = __expf(h[k] - m); s += ex[k]; }
  const float inv = 1.f / s;
  float* pi = out;
  float* mu = out + (size_t)N * 20;
  float* ls = out + (size_t)2 * N * 20;
#pragma unroll
  for (int k = 0; k < 20; ++k) {
    pi[(size_t)i * 20 + k] = ex[k] * inv;
    mu[(size_t)i * 20 + k] = h[20 + k];
    ls[(size_t)i * 20 + k] = h[40 + k];
  }
}

// ---------------------------------------------------------------------------
extern "C" void kernel_launch(void* const* d_in, const int* in_sizes, int n_in,
                              void* d_out, int out_size, void* d_ws,
                              size_t ws_size, hipStream_t stream) {
  const float* x   = (const float*)d_in[0];
  const int*   ei  = (const int*)d_in[1];
  const float* W1  = (const float*)d_in[2];
  const float* b1  = (const float*)d_in[3];
  const float* W2  = (const float*)d_in[4];
  const float* b2  = (const float*)d_in[5];
  const float* piW = (const float*)d_in[6];
  const float* pib = (const float*)d_in[7];
  const float* muW = (const float*)d_in[8];
  const float* mub = (const float*)d_in[9];
  const float* lsW = (const float*)d_in[10];
  const float* lsb = (const float*)d_in[11];

  const int N = in_sizes[0] / 128;
  const int E = in_sizes[1] / 2;
  const int* src = ei;
  const int* dst = ei + E;
  const int NF = N * 128;

  // Workspace layout (f32 unless noted)
  float* dis  = (float*)d_ws;                      // [N]
  float* bufH = dis + N;                           // [N,128] GEMM outputs
  float* bufA = bufH + (size_t)NF;                 // [N,128] agg1 -> h1(relu)
  float* bufB = bufA + (size_t)NF;                 // [N,128] agg2 -> h2(relu)
  unsigned short* Wt1 = (unsigned short*)(bufB + (size_t)NF); // bf16 [128][128]
  unsigned short* Wt2 = Wt1 + 128 * 128;
  unsigned short* Wht = Wt2 + 128 * 128;           // bf16 [64][128]
  float* bh = (float*)(Wht + 64 * 128);            // [64]

  const int gemmGrid = (N + 63) / 64;
  const int nfGrid   = (NF + 255) / 256;
  const int nGrid    = (N + 255) / 256;
  const int eGrid    = (E + 255) / 256;
  const int sGrid    = (E + 7) / 8;                // 8 edges (waves) per block

  prep_weights<<<(41024 + 255) / 256, 256, 0, stream>>>(
      W1, W2, piW, muW, lsW, pib, mub, lsb, Wt1, Wt2, Wht, bh);

  // degree -> dis = 1/sqrt(deg+1)
  zero_f32<<<nGrid, 256, 0, stream>>>(dis, N);
  edge_deg<<<eGrid, 256, 0, stream>>>(dst, dis, E);
  inv_sqrt_deg<<<nGrid, 256, 0, stream>>>(dis, N);

  // ---- Layer 1 ----
  gemm_bf16_wmma<128><<<gemmGrid, 128, 0, stream>>>(x, Wt1, nullptr, bufH, N);
  zero_f32<<<nfGrid, 256, 0, stream>>>(bufA, NF);
  scatter_edges<<<sGrid, 256, 0, stream>>>(src, dst, bufH, dis, bufA, E);
  finalize_relu<<<nfGrid, 256, 0, stream>>>(bufA, bufH, dis, b1, N);

  // ---- Layer 2 ----
  gemm_bf16_wmma<128><<<gemmGrid, 128, 0, stream>>>(bufA, Wt2, nullptr, bufH, N);
  zero_f32<<<nfGrid, 256, 0, stream>>>(bufB, NF);
  scatter_edges<<<sGrid, 256, 0, stream>>>(src, dst, bufH, dis, bufB, E);
  finalize_relu<<<nfGrid, 256, 0, stream>>>(bufB, bufH, dis, b2, N);

  // ---- MDN heads ----
  gemm_bf16_wmma<64><<<gemmGrid, 128, 0, stream>>>(bufB, Wht, bh, bufH, N);
  heads_out<<<nGrid, 256, 0, stream>>>(bufH, (float*)d_out, N);
}
